// CoarseEncoder_vssm_48936857371355
// MI455X (gfx1250) — compile-verified, matrix operand order checked by reading, not run on hardware
//
#include <hip/hip_runtime.h>
#include <hip/hip_bf16.h>
#include <math.h>

typedef __attribute__((ext_vector_type(16))) _Float16 v16h;
typedef __attribute__((ext_vector_type(8)))  float    v8f;
typedef __attribute__((ext_vector_type(4)))  unsigned int ui4;
typedef __attribute__((ext_vector_type(8)))  int      i8v;
typedef __attribute__((ext_vector_type(4)))  int      i4v;

#define BB   2
#define CC   96
#define HH   64
#define WW   256
#define LL   (HH*WW)          /* 16384 */
#define KK   4
#define RR   6
#define NLYR 2
#define SEG  (LL/256)         /* 64 */
#define SZ   ((size_t)BB*CC*LL)

#if defined(__HIP_DEVICE_COMPILE__) && __has_builtin(__builtin_amdgcn_tensor_load_to_lds)
#define HAS_TDM 1
#else
#define HAS_TDM 0
#endif

// scan index -> row-major pixel index, per direction k
__device__ __forceinline__ int scan_pix(int k, int j) {
  switch (k & 3) {
    case 0:  return j;                                         // row-major
    case 1:  return ((j & (HH-1)) * WW) + (j >> 6);            // col-major
    case 2:  return (LL - 1) - j;                              // row reversed
    default: { int m = (LL - 1) - j;                           // col reversed
               return ((m & (HH-1)) * WW) + (m >> 6); }
  }
}

// ---------------- concat + reversed stream ----------------
__global__ __launch_bounds__(256) void k_concat(const float* __restrict__ x0,
                                                const float* __restrict__ x1,
                                                float* __restrict__ xA,
                                                float* __restrict__ xB) {
  size_t idx = (size_t)blockIdx.x * 256 + threadIdx.x;
  if (idx >= SZ) return;
  int wf = idx % WW;
  int h  = (idx / WW) % HH;
  int c  = (idx / LL) % CC;
  int b  = idx / ((size_t)CC * LL);
  float v = (wf < 128) ? x0[(((size_t)(b*CC + c))*HH + h)*128 + wf]
                       : x1[(((size_t)(b*CC + c))*HH + h)*128 + (wf - 128)];
  size_t base = (size_t)(b*CC + c) * LL + (size_t)h * WW;
  xA[base + wf]             = v;
  xB[base + (WW - 1 - wf)]  = v;
}

// ---------------- per-pixel channel LayerNorm ----------------
__global__ __launch_bounds__(256) void k_ln2d(const float* __restrict__ X,
                                              float* __restrict__ Y,
                                              const float* __restrict__ gw,
                                              const float* __restrict__ gb) {
  int idx = blockIdx.x * 256 + threadIdx.x;       // over BB*LL pixels
  if (idx >= BB * LL) return;
  int b = idx / LL, l = idx % LL;
  const float* xp = X + (size_t)b * CC * LL + l;
  float s = 0.f, s2 = 0.f;
  for (int c = 0; c < CC; ++c) {
    float v = xp[(size_t)c * LL];
    s += v; s2 += v * v;
  }
  float mu  = s  * (1.f / CC);
  float var = s2 * (1.f / CC) - mu * mu;
  float inv = rsqrtf(var + 1e-5f);
  float* yp = Y + (size_t)b * CC * LL + l;
  for (int c = 0; c < CC; ++c)
    yp[(size_t)c * LL] = (xp[(size_t)c * LL] - mu) * inv * gw[c] + gb[c];
}

// ---------------- WMMA GEMM, TDM -> LDS -> WMMA path ------------------------
// Block (256 thr / 8 waves) owns a 32(M) x 64(N) output tile of
// Y[b,m,n] = act(W[m,:]·X[b,:,n] + bias) (optionally += into Y).
// Per K-step of 32: one tensor_load_to_lds DMAs the 32x64 f32 X tile into LDS
// (tile_dim0=64, tile_dim1=32, dim0_stride=LL), s_wait_tensorcnt + barrier,
// then each wave runs one v_wmma_f32_16x16x32_f16 (K loop fully unrolled).
template<int M, int K, int ACT, int RESID>
__global__ __launch_bounds__(256) void k_gemm(const float* __restrict__ Wm,
                                              const float* __restrict__ X,
                                              float* __restrict__ Y,
                                              const float* __restrict__ bias) {
  __shared__ float xtile[32 * 64];
  const int wave = threadIdx.x >> 5;
  const int lane = threadIdx.x & 31;

  constexpr int nBlkN    = LL / 64;
  constexpr int nBlkPerB = (M / 32) * nBlkN;
  const int b   = blockIdx.x / nBlkPerB;
  const int t   = blockIdx.x % nBlkPerB;
  const int mB0 = (t / nBlkN) * 32;
  const int nB0 = (t % nBlkN) * 64;

  const int m0   = mB0 + (wave >> 2) * 16;     // wave's 16x16 sub-tile
  const int nloc = (wave & 3) * 16;

  const int mA = lane & 15;                    // A frag: 16x32 f16 layout
  const int kB = (lane >> 4) ? 8 : 0;
  const int nB = lane & 15;

  const float* wrow  = Wm + (size_t)(m0 + mA) * K;
  const float* xbase = X + (size_t)b * K * LL + nB0;

  v8f acc = {};

#pragma unroll
  for (int k0 = 0; k0 < K; k0 += 32) {
    __syncthreads();                           // previous tile fully consumed
#if HAS_TDM
    if (wave == 0) {
      unsigned long long ga = (unsigned long long)(uintptr_t)(xbase + (size_t)k0 * LL);
      unsigned ldsa = (unsigned)(uintptr_t)&xtile[0];
      // D# group0: count=1 | lds_addr | global_addr[56:0] | type=2
      ui4 g0 = { 1u, ldsa, (unsigned)(ga & 0xffffffffu),
                 (unsigned)((ga >> 32) & 0x1ffffffu) | (2u << 30) };
      // D# group1: data_size=4B; tensor_dim0=LL; tensor_dim1=K;
      //            tile_dim0=64; tile_dim1=32; dim0_stride=LL
      i8v g1 = { 0x00020000, (int)((LL & 0xffff) << 16), (int)(K << 16),
                 (int)(64 << 16), 32, (int)LL, 0, 0 };
      i4v z4 = { 0, 0, 0, 0 };
#if __clang_major__ >= 23
      i8v z8 = { 0, 0, 0, 0, 0, 0, 0, 0 };
      __builtin_amdgcn_tensor_load_to_lds(g0, g1, z4, z4, z8, 0);
#else
      __builtin_amdgcn_tensor_load_to_lds(g0, g1, z4, z4, 0);
#endif
      __builtin_amdgcn_s_wait_tensorcnt(0);
    }
#else
    for (int i = threadIdx.x; i < 32 * 64; i += 256)
      xtile[i] = xbase[(size_t)((i >> 6) + k0) * LL + (i & 63)];
#endif
    __syncthreads();                           // tile visible to all waves

    v16h afrag, bfrag;
#pragma unroll
    for (int e = 0; e < 16; ++e) {
      int v = e >> 1, hh = e & 1;
      int kk = kB + ((v < 4) ? (v * 2 + hh) : (16 + (v - 4) * 2 + hh));
      afrag[e] = (_Float16)wrow[k0 + kk];
      bfrag[e] = (_Float16)xtile[kk * 64 + nloc + nB];
    }
    acc = __builtin_amdgcn_wmma_f32_16x16x32_f16(false, afrag, false, bfrag,
                                                 (short)0, acc, false, false);
  }

  const int mBase = (lane >> 4) * 8;           // D: vgpr r -> M = r + 8*(lane>>4)
#pragma unroll
  for (int r = 0; r < 8; ++r) {
    int m = m0 + mBase + r;
    float v = acc[r];
    if (bias) v += bias[m];
    if (ACT == 1)      v = v / (1.f + __expf(-v));                     // SiLU
    else if (ACT == 2) v = 0.5f * v * (1.f + erff(v * 0.70710678f));   // GELU
    float* yp = Y + (size_t)(b * M + m) * LL + nB0 + nloc + nB;
    if (RESID) *yp += v; else *yp = v;
  }
}

// ---------------- depthwise 3x3 SAME + SiLU ----------------
__global__ __launch_bounds__(256) void k_dwconv_silu(const float* __restrict__ X,
                                                     float* __restrict__ Y,
                                                     const float* __restrict__ w3) {
  size_t idx = (size_t)blockIdx.x * 256 + threadIdx.x;
  if (idx >= SZ) return;
  int w = idx % WW;
  int h = (idx / WW) % HH;
  int c = (idx / LL) % CC;
  int b = idx / ((size_t)CC * LL);
  const float* xp = X + (size_t)(b * CC + c) * LL;
  const float* wp = w3 + c * 9;
  float s = 0.f;
#pragma unroll
  for (int dy = -1; dy <= 1; ++dy) {
    int hy = h + dy; if ((unsigned)hy >= (unsigned)HH) continue;
#pragma unroll
    for (int dx = -1; dx <= 1; ++dx) {
      int wx = w + dx; if ((unsigned)wx >= (unsigned)WW) continue;
      s += xp[(size_t)hy * WW + wx] * wp[(dy + 1) * 3 + (dx + 1)];
    }
  }
  Y[idx] = s / (1.f + __expf(-s));
}

// ---------------- x_proj: xdbl[b,k,d,l] = sum_c w[k,d,c]*xs[b,k,c,l] --------
__global__ __launch_bounds__(256) void k_xproj(const float* __restrict__ xact,
                                               const float* __restrict__ xpw,
                                               float* __restrict__ xdbl) {
  int idx = blockIdx.x * 256 + threadIdx.x;       // BB*KK*LL
  if (idx >= BB * KK * LL) return;
  int j = idx % LL;
  int k = (idx / LL) % KK;
  int b = idx / (KK * LL);
  const float* xp = xact + (size_t)b * CC * LL + scan_pix(k, j);
  const float* wp = xpw + (size_t)k * (RR + 2) * CC;
  float acc[RR + 2];
#pragma unroll
  for (int d = 0; d < RR + 2; ++d) acc[d] = 0.f;
  for (int c = 0; c < CC; ++c) {
    float xv = xp[(size_t)c * LL];
#pragma unroll
    for (int d = 0; d < RR + 2; ++d) acc[d] += wp[d * CC + c] * xv;
  }
  float* op = xdbl + (size_t)(b * KK + k) * (RR + 2) * LL + j;
#pragma unroll
  for (int d = 0; d < RR + 2; ++d) op[(size_t)d * LL] = acc[d];
}

__global__ __launch_bounds__(256) void k_zero(float* __restrict__ Y) {
  size_t idx = (size_t)blockIdx.x * 256 + threadIdx.x;
  if (idx < SZ) Y[idx] = 0.f;
}

// ---------------- selective scan, 2 directions per block --------------------
// block = (b,c,pair); 256 threads, SEG=64-elt segments, LDS carry scan,
// 4-direction accumulation via global_atomic_add_f32 into Y (pre-zeroed).
__global__ __launch_bounds__(256) void k_scan(const float* __restrict__ xact,
                                              const float* __restrict__ xdbl,
                                              const float* __restrict__ dtw,
                                              const float* __restrict__ dtb,
                                              const float* __restrict__ alog,
                                              const float* __restrict__ dsv,
                                              float* __restrict__ Y) {
  __shared__ float sA[256];
  __shared__ float sB[256];
  const int tid  = threadIdx.x;
  const int pair = blockIdx.x & 1;           // 0: row dirs {0,2}, 1: col dirs {1,3}
  const int bc   = blockIdx.x >> 1;
  const int c    = bc % CC;
  const int b    = bc / CC;
  const int j0   = tid * SEG;

  for (int dir = 0; dir < 2; ++dir) {
    const int k = pair + 2 * dir;
    float w0 = dtw[(k * CC + c) * RR + 0], w1 = dtw[(k * CC + c) * RR + 1];
    float w2 = dtw[(k * CC + c) * RR + 2], w3 = dtw[(k * CC + c) * RR + 3];
    float w4 = dtw[(k * CC + c) * RR + 4], w5 = dtw[(k * CC + c) * RR + 5];
    float bdt = dtb[k * CC + c];
    float Ac  = -__expf(alog[k * CC + c]);
    float Dch = dsv[k * CC + c];
    const float* xd   = xdbl + (size_t)(b * KK + k) * (RR + 2) * LL;
    const float* xrow = xact + (size_t)(b * CC + c) * LL;

    // phase 1: segment aggregate (Aag, Bag)
    float Aag = 1.f, Bag = 0.f;
    for (int j = j0; j < j0 + SEG; ++j) {
      float pre = w0 * xd[0 * LL + j] + w1 * xd[1 * LL + j] + w2 * xd[2 * LL + j]
                + w3 * xd[3 * LL + j] + w4 * xd[4 * LL + j] + w5 * xd[5 * LL + j] + bdt;
      float delta = (pre > 20.f) ? pre : __logf(1.f + __expf(pre));
      float a  = __expf(delta * Ac);
      float xv = xrow[scan_pix(k, j)];
      float bv = delta * xd[6 * LL + j] * xv;
      Aag = a * Aag;
      Bag = a * Bag + bv;
    }
    sA[tid] = Aag; sB[tid] = Bag;
    __syncthreads();
    // inclusive Hillis-Steele scan over (a,b) pairs in LDS
    for (int off = 1; off < 256; off <<= 1) {
      float pa = 1.f, pb = 0.f;
      if (tid >= off) { pa = sA[tid - off]; pb = sB[tid - off]; }
      __syncthreads();
      float na = sA[tid] * pa;
      float nb = sA[tid] * pb + sB[tid];
      __syncthreads();
      sA[tid] = na; sB[tid] = nb;
      __syncthreads();
    }
    float h = (tid == 0) ? 0.f : sB[tid - 1];    // exclusive carry (h_init = 0)

    // phase 2: replay with carry, scatter ys into accumulator
    for (int j = j0; j < j0 + SEG; ++j) {
      float pre = w0 * xd[0 * LL + j] + w1 * xd[1 * LL + j] + w2 * xd[2 * LL + j]
                + w3 * xd[3 * LL + j] + w4 * xd[4 * LL + j] + w5 * xd[5 * LL + j] + bdt;
      float delta = (pre > 20.f) ? pre : __logf(1.f + __expf(pre));
      float a  = __expf(delta * Ac);
      int   px = scan_pix(k, j);
      float xv = xrow[px];
      h = a * h + delta * xd[6 * LL + j] * xv;
      float ys = xd[7 * LL + j] * h + Dch * xv;
      atomicAdd(&Y[(size_t)(b * CC + c) * LL + px], ys);
    }
    __syncthreads();
  }
}

// ---------------- final: y = 0.5*(x + reverse(xb)), split back to y0|y1 -----
__global__ __launch_bounds__(256) void k_final(const float* __restrict__ xA,
                                               const float* __restrict__ xB,
                                               float* __restrict__ out) {
  size_t idx = (size_t)blockIdx.x * 256 + threadIdx.x;
  if (idx >= SZ) return;
  const size_t half = SZ / 2;                 // elements in y0
  int which = (int)(idx / half);
  size_t r = idx % half;
  int w  = r % 128;
  int h  = (r / 128) % HH;
  int c  = (r / (128 * HH)) % CC;
  int b  = r / ((size_t)128 * HH * CC);
  int wf = which * 128 + w;
  size_t base = (size_t)(b * CC + c) * LL + (size_t)h * WW;
  out[idx] = 0.5f * (xA[base + wf] + xB[base + (WW - 1 - wf)]);
}

extern "C" void kernel_launch(void* const* d_in, const int* in_sizes, int n_in,
                              void* d_out, int out_size, void* d_ws, size_t ws_size,
                              hipStream_t stream) {
  (void)in_sizes; (void)n_in; (void)out_size; (void)ws_size;
  const float* x0        = (const float*)d_in[0];
  const float* x1        = (const float*)d_in[1];
  const float* norm_w    = (const float*)d_in[2];
  const float* norm_b    = (const float*)d_in[3];
  const float* in_proj_w = (const float*)d_in[4];
  const float* conv_w    = (const float*)d_in[5];
  const float* x_proj_w  = (const float*)d_in[6];
  const float* dt_w      = (const float*)d_in[7];
  const float* dt_b      = (const float*)d_in[8];
  const float* A_logs    = (const float*)d_in[9];
  const float* Ds        = (const float*)d_in[10];
  const float* onw       = (const float*)d_in[11];
  const float* onb       = (const float*)d_in[12];
  const float* out_proj_w= (const float*)d_in[13];
  const float* norm2_w   = (const float*)d_in[14];
  const float* norm2_b   = (const float*)d_in[15];
  const float* fc1_w     = (const float*)d_in[16];
  const float* fc1_b     = (const float*)d_in[17];
  const float* fc2_w     = (const float*)d_in[18];
  const float* fc2_b     = (const float*)d_in[19];

  float* ws = (float*)d_ws;
  float* xS[2] = { ws, ws + SZ };                 // the two streams
  float* t0   = ws + 2 * SZ;                      // ln out / conv act (xs source)
  float* t1   = ws + 3 * SZ;                      // in_proj out / scan accumulator
  float* t2   = ws + 4 * SZ;                      // out-ln / mlp-ln
  float* hid  = ws + 5 * SZ;                      // MLP hidden, 4*SZ floats
  float* xdbl = ws + 9 * SZ;                      // (B,K,8,L)

  const int elGrid = (int)((SZ + 255) / 256);     // 12288
  const int lnGrid = (BB * LL + 255) / 256;       // 128
  const int g96    = BB * (96  / 32) * (LL / 64); // 1536 blocks, 32x64 tiles
  const int g384   = BB * (384 / 32) * (LL / 64); // 6144

  k_concat<<<elGrid, 256, 0, stream>>>(x0, x1, xS[0], xS[1]);

  for (int s = 0; s < 2; ++s) {
    float* x = xS[s];
    for (int i = 0; i < NLYR; ++i) {
      // --- ss2d ---
      k_ln2d<<<lnGrid, 256, 0, stream>>>(x, t0, norm_w + i * CC, norm_b + i * CC);
      k_gemm<96, 96, 0, 0><<<g96, 256, 0, stream>>>(
          in_proj_w + (size_t)i * CC * CC, t0, t1, nullptr);
      k_dwconv_silu<<<elGrid, 256, 0, stream>>>(t1, t0, conv_w + (size_t)i * CC * 9);
      k_xproj<<<(BB * KK * LL) / 256, 256, 0, stream>>>(
          t0, x_proj_w + (size_t)i * KK * (RR + 2) * CC, xdbl);
      k_zero<<<elGrid, 256, 0, stream>>>(t1);
      k_scan<<<BB * CC * 2, 256, 0, stream>>>(
          t0, xdbl, dt_w + (size_t)i * KK * CC * RR, dt_b + (size_t)i * KK * CC,
          A_logs + (size_t)i * KK * CC, Ds + (size_t)i * KK * CC, t1);
      k_ln2d<<<lnGrid, 256, 0, stream>>>(t1, t2, onw + i * CC, onb + i * CC);
      k_gemm<96, 96, 0, 1><<<g96, 256, 0, stream>>>(
          out_proj_w + (size_t)i * CC * CC, t2, x, nullptr);     // residual
      // --- MLP ---
      k_ln2d<<<lnGrid, 256, 0, stream>>>(x, t2, norm2_w + i * CC, norm2_b + i * CC);
      k_gemm<384, 96, 2, 0><<<g384, 256, 0, stream>>>(
          fc1_w + (size_t)i * 4 * CC * CC, t2, hid, fc1_b + (size_t)i * 4 * CC);
      k_gemm<96, 384, 0, 1><<<g96, 256, 0, stream>>>(
          fc2_w + (size_t)i * CC * 4 * CC, hid, x, fc2_b + (size_t)i * CC);
    }
  }

  k_final<<<elGrid, 256, 0, stream>>>(xS[0], xS[1], (float*)d_out);
}
